// MultiHeadAttention_30966714204414
// MI455X (gfx1250) — matrix operand order repaired and checked by reading, not compiled, not run on hardware
//
#include <hip/hip_runtime.h>
#include <hip/hip_bf16.h>
#include <stdint.h>

// Problem constants (B=2, S=2048, H=1024, NH=16, HD=64)
#define S_LEN  2048
#define HDIM   1024
#define NHEADS 16
#define HEADD  64
#define BATCH  2
#define MROWS  (BATCH * S_LEN)   // 4096

typedef __attribute__((ext_vector_type(16))) __bf16 bf16x16;
typedef __attribute__((ext_vector_type(8)))  float  v8f;

union FragAB {
  bf16x16 v;
  uint4   q[2];
};

__device__ __forceinline__ unsigned short f2bf(float f) {
  union { float f; uint32_t u; } c; c.f = f;
  uint32_t u = c.u;
  uint32_t r = u + 0x7fffu + ((u >> 16) & 1u);   // round-to-nearest-even
  if ((u & 0x7f800000u) == 0x7f800000u) r = u;   // inf/nan passthrough
  return (unsigned short)(r >> 16);
}

__device__ __forceinline__ v8f zero8() {
  v8f z;
#pragma unroll
  for (int i = 0; i < 8; ++i) z[i] = 0.0f;
  return z;
}

__device__ __forceinline__ v8f wmma_bf16(const FragAB& a, const FragAB& b, v8f c) {
  return __builtin_amdgcn_wmma_f32_16x16x32_bf16(
      /*neg_a=*/false, a.v, /*neg_b=*/false, b.v,
      /*c_mod=*/(short)0, c, /*reuse_a=*/false, /*reuse_b=*/false);
}

// Async global->LDS 16-byte copy (CDNA5 GLOBAL_LOAD_ASYNC_TO_LDS_B128, ASYNCcnt).
// lds_off = workgroup-relative LDS byte address (low 32 bits of the generic pointer).
__device__ __forceinline__ void async_g2l_b128(uint32_t lds_off, const void* gaddr) {
  asm volatile("global_load_async_to_lds_b128 %0, %1, off"
               :: "v"(lds_off), "v"(gaddr) : "memory");
}
__device__ __forceinline__ uint32_t lds_off32(const void* p) {
  return (uint32_t)(size_t)p;          // LDS aperture: addr[31:0] = LDS byte offset
}
#define S_WAIT_ASYNC(n) asm volatile("s_wait_asynccnt " #n ::: "memory")

// ---------------- pack kernels: fp32 -> bf16 ----------------

__global__ __launch_bounds__(256) void pack_x_kernel(
    const float* __restrict__ src, unsigned short* __restrict__ dst, int n) {
  int i = blockIdx.x * 256 + threadIdx.x;
  if (i < n) dst[i] = f2bf(src[i]);
}

// W [K=1024, N=1024] fp32 row-major  ->  Wt [N, K] bf16 (so B-fragment loads are contiguous)
__global__ __launch_bounds__(256) void pack_wt_kernel(
    const float* __restrict__ W, unsigned short* __restrict__ Wt) {
  int i = blockIdx.x * 256 + threadIdx.x;   // i < 1024*1024
  int n = i >> 10;
  int k = i & 1023;
  Wt[(size_t)n * HDIM + k] = f2bf(W[(size_t)k * HDIM + n]);
}

// ---------------- bf16 GEMM: C[4096x1024] = A[4096x1024] * Wt^T + bias ----------------
// Block = 8 waves = 256 M-rows x 64 N-cols; each wave owns a 32(M) x 64(N) tile
// (2 A-frags, 8 accumulators) so every LDS B-fragment feeds TWO wmma ops.
// The 64x32 B tile (4 KB) is shared by all 8 waves: fetched once per block per k-step via
// cooperative global_load_async_to_lds_b128 (double-buffered, one tile in flight; last
// iteration peeled so the steady-state loop is a single uniform block).
//
// mode 0: store bf16 split-head [B, NH, S, HD]            (Q, K)
// mode 1: store bf16 split-head TRANSPOSED [B, NH, HD, S] (V)
// mode 2: store fp32 row-major [M, N]                     (final output)
__global__ __launch_bounds__(256) void gemm_bf16_kernel(
    const unsigned short* __restrict__ A,   // [4096 x 1024] bf16 row-major
    const unsigned short* __restrict__ Bt,  // [1024 x 1024] bf16, row n = weight column n
    const float* __restrict__ bias,         // [1024]
    void* __restrict__ dst, int mode) {
  __shared__ __align__(16) unsigned short btile[2][64 * 32];   // 8 KB double buffer

  const int tid  = threadIdx.x;
  const int lane = tid & 31;
  const int wave = tid >> 5;
  const int half = lane >> 4;
  const int l16  = lane & 15;
  const int m0 = (blockIdx.y * 8 + wave) * 32;
  const int n0 = blockIdx.x * 64;

  // cooperative B-tile fetch: 256 threads x 16B = 4 KB tile; row r=tid>>2, chunk (tid&3)*8
  const int br = tid >> 2;
  const int bc = (tid & 3) * 8;
  const unsigned short* bsrc = Bt + (size_t)(n0 + br) * HDIM + bc;

  v8f acc[8];
#pragma unroll
  for (int t = 0; t < 8; ++t) acc[t] = zero8();

  const unsigned short* aptr0 = A + (size_t)(m0 + l16) * HDIM;
  const unsigned short* aptr1 = aptr0 + (size_t)16 * HDIM;

  auto compute = [&](int kt) {
    const int k0 = kt * 32;
    const int ak = k0 + half * 8;          // A pattern: {0..7,16..23} / {8..15,24..31}
    FragAB a0, a1;
    a0.q[0] = *(const uint4*)(aptr0 + ak);
    a0.q[1] = *(const uint4*)(aptr0 + ak + 16);
    a1.q[0] = *(const uint4*)(aptr1 + ak);
    a1.q[1] = *(const uint4*)(aptr1 + ak + 16);
    const unsigned short* bt = &btile[kt & 1][0];
#pragma unroll
    for (int t = 0; t < 4; ++t) {
      FragAB b;                            // B pattern: {0..7,16..23} / {8..15,24..31}
      const unsigned short* bp = bt + (t * 16 + l16) * 32 + half * 8;
      b.q[0] = *(const uint4*)(bp);        // ds_load_b128
      b.q[1] = *(const uint4*)(bp + 16);
      acc[t]     = wmma_bf16(a0, b, acc[t]);      // B-fragment reused for both M halves
      acc[4 + t] = wmma_bf16(a1, b, acc[4 + t]);
    }
  };

  async_g2l_b128(lds_off32(&btile[0][br * 32 + bc]), bsrc);            // prologue: tile 0
  for (int kt = 0; kt < 31; ++kt) {        // uniform steady-state loop
    async_g2l_b128(lds_off32(&btile[(kt + 1) & 1][br * 32 + bc]),
                   bsrc + (size_t)(kt + 1) * 32);
    S_WAIT_ASYNC(0x1);                     // current tile done (one prefetch in flight)
    __syncthreads();
    compute(kt);
    __syncthreads();                       // all waves done with this buffer
  }
  S_WAIT_ASYNC(0x0);                       // peeled last iteration
  __syncthreads();
  compute(31);

  // C/D layout: VGPR i -> row mbase + i + 8*half, column n0 + t*16 + l16
#pragma unroll
  for (int g = 0; g < 2; ++g) {
    const int mbase = m0 + g * 16;
    const int b  = mbase >> 11;            // batch index (tiles never cross batch)
    const int sB = mbase & (S_LEN - 1);
#pragma unroll
    for (int t = 0; t < 4; ++t) {
      const v8f& ac = acc[g * 4 + t];
      const int ncol = n0 + t * 16 + l16;
      const float bia = bias[ncol];
      const int head = ncol >> 6;
      const int d    = ncol & 63;
      if (mode == 0) {          // [B, NH, S, HD] bf16
        unsigned short* o = (unsigned short*)dst;
#pragma unroll
        for (int i = 0; i < 8; ++i) {
          const int s = sB + i + half * 8;
          o[(((size_t)b * NHEADS + head) * S_LEN + s) * HEADD + d] = f2bf(ac[i] + bia);
        }
      } else if (mode == 1) {   // [B, NH, HD, S] bf16 (V transposed) -> contiguous 16B store
        unsigned short* o = (unsigned short*)dst;
        unsigned short tmp[8];
#pragma unroll
        for (int i = 0; i < 8; ++i) tmp[i] = f2bf(ac[i] + bia);
        const size_t base = (((size_t)b * NHEADS + head) * HEADD + d) * S_LEN + sB + half * 8;
        *(uint4*)(o + base) = *(const uint4*)tmp;
      } else {                  // fp32 row-major [M, 1024]
        float* o = (float*)dst;
#pragma unroll
        for (int i = 0; i < 8; ++i) {
          const int m = mbase + i + half * 8;
          o[(size_t)m * HDIM + ncol] = ac[i] + bia;
        }
      }
    }
  }
}

// ---------------- fused flash-style attention ----------------
// Block = 8 waves, all on the same (b,h); wave w owns a 16-query tile. The 32-key K tile
// (32x64, 4 KB) and V tile (64x32, 4 KB) are shared by all 8 waves: async-copied to LDS
// once per block per step (double-buffered, last step peeled), fragments read back with
// ds_load_b128. Per step: 4 wmma (scores) -> scale+mask -> online softmax (shfl-xor in
// 16-lane halves) -> P via per-wave LDS transpose -> 4 wmma (P*V).
__global__ __launch_bounds__(256) void attn_kernel(
    const unsigned short* __restrict__ Q,   // [B, NH, S, HD] bf16
    const unsigned short* __restrict__ Kk,  // [B, NH, S, HD] bf16
    const unsigned short* __restrict__ Vt,  // [B, NH, HD, S] bf16
    const float* __restrict__ mask,         // [B, S], 1.0 -> add -10000
    unsigned short* __restrict__ ctx) {     // [B, S, H] bf16
  __shared__ __align__(16) unsigned short ktile[2][32 * 64];   // 8 KB
  __shared__ __align__(16) unsigned short vtile[2][64 * 32];   // 8 KB
  __shared__ __align__(16) unsigned short pbuf[8][16 * 32];    // 8 KB per-wave P bounce

  const int tid  = threadIdx.x;
  const int lane = tid & 31;
  const int wave = tid >> 5;
  const int half = lane >> 4;
  const int l16  = lane & 15;

  const int bh = blockIdx.y;               // 0..31
  const int b  = bh >> 4;
  const int h  = bh & 15;
  const int q0 = (blockIdx.x * 8 + wave) * 16;

  const unsigned short* Qb = Q  + ((size_t)bh * S_LEN + q0) * HEADD;
  const unsigned short* Kb = Kk + (size_t)bh * S_LEN * HEADD;
  const unsigned short* Vb = Vt + (size_t)bh * HEADD * S_LEN;
  const float*          mb = mask + (size_t)b * S_LEN;

  // cooperative tile mapping (256 threads x 16B = 4 KB each):
  const int kr = tid >> 3, kc = (tid & 7) * 8;   // K tile [32 keys][64 d]
  const int vr = tid >> 2, vc = (tid & 3) * 8;   // V tile [64 d][32 keys]

  // Q fragments: rows q0..q0+15, d 0..31 and 32..63 (loaded once, direct)
  FragAB qa[2];
  {
    const unsigned short* qp = Qb + (size_t)l16 * HEADD;
#pragma unroll
    for (int t = 0; t < 2; ++t) {
      const int dk = t * 32 + half * 16;
      qa[t].q[0] = *(const uint4*)(qp + dk);
      qa[t].q[1] = *(const uint4*)(qp + dk + 8);
    }
  }

  v8f o[4];
#pragma unroll
  for (int t = 0; t < 4; ++t) o[t] = zero8();
  float mrow[8], lrow[8];
#pragma unroll
  for (int i = 0; i < 8; ++i) { mrow[i] = -1.0e30f; lrow[i] = 0.0f; }

  auto compute = [&](int st) {
    const int kt  = st * 32;
    const int cur = st & 1;

    // scores: s0 = Q @ K[kt..kt+15]^T, s1 = Q @ K[kt+16..kt+31]^T  (inner dim HD = 64)
    v8f s0 = zero8(), s1 = zero8();
    {
      const unsigned short* kp0 = &ktile[cur][(size_t)l16 * 64];        // key l16
      const unsigned short* kp1 = &ktile[cur][(size_t)(16 + l16) * 64]; // key 16+l16
#pragma unroll
      for (int t = 0; t < 2; ++t) {
        const int dk = t * 32 + half * 16;
        FragAB kb;
        kb.q[0] = *(const uint4*)(kp0 + dk);
        kb.q[1] = *(const uint4*)(kp0 + dk + 8);
        s0 = wmma_bf16(qa[t], kb, s0);
        kb.q[0] = *(const uint4*)(kp1 + dk);
        kb.q[1] = *(const uint4*)(kp1 + dk + 8);
        s1 = wmma_bf16(qa[t], kb, s1);
      }
    }

    const float ma0 = mb[kt + l16]      * -10000.0f;
    const float ma1 = mb[kt + 16 + l16] * -10000.0f;

    float al[8];
#pragma unroll
    for (int i = 0; i < 8; ++i) {
      float x0 = s0[i] * 0.125f + ma0;     // 1/sqrt(64)
      float x1 = s1[i] * 0.125f + ma1;
      float t = fmaxf(x0, x1);             // row max across 16 columns (one C-frag row
#pragma unroll
      for (int m = 1; m <= 8; m <<= 1)     //  lives in lanes 0..15 / 16..31 of one VGPR)
        t = fmaxf(t, __shfl_xor(t, m, 32));
      const float nm = fmaxf(mrow[i], t);
      const float p0 = __expf(x0 - nm);
      const float p1 = __expf(x1 - nm);
      s0[i] = p0; s1[i] = p1;
      float rs = p0 + p1;
#pragma unroll
      for (int m = 1; m <= 8; m <<= 1) rs += __shfl_xor(rs, m, 32);
      al[i]   = __expf(mrow[i] - nm);
      lrow[i] = lrow[i] * al[i] + rs;
      mrow[i] = nm;
    }
#pragma unroll
    for (int t = 0; t < 4; ++t)
#pragma unroll
      for (int i = 0; i < 8; ++i) o[t][i] *= al[i];

    // P (C-fragment layout) -> per-wave LDS tile -> A-fragment layout, as bf16
#pragma unroll
    for (int i = 0; i < 8; ++i) {
      const int row = i + half * 8;
      pbuf[wave][row * 32 + l16]      = f2bf(s0[i]);
      pbuf[wave][row * 32 + 16 + l16] = f2bf(s1[i]);
    }
    __asm__ volatile("s_wait_dscnt 0x0" ::: "memory");   // per-wave LDS RAW fence

    FragAB pa;
    {
      const unsigned short* pp = &pbuf[wave][(size_t)l16 * 32];
      const int kk = half * 16;
      pa.q[0] = *(const uint4*)(pp + kk);
      pa.q[1] = *(const uint4*)(pp + kk + 8);
    }

    // O += P[16x32] @ V[32x64]  (V tile stored d-major: contiguous B-frag ds loads)
#pragma unroll
    for (int t = 0; t < 4; ++t) {
      FragAB vbf;
      const unsigned short* vp = &vtile[cur][(size_t)(t * 16 + l16) * 32 + half * 16];
      vbf.q[0] = *(const uint4*)(vp);
      vbf.q[1] = *(const uint4*)(vp + 8);
      o[t] = wmma_bf16(pa, vbf, o[t]);
    }
  };

  // prologue: async-fill tile 0 (2 loads per thread: K chunk + V chunk)
  async_g2l_b128(lds_off32(&ktile[0][kr * 64 + kc]), Kb + (size_t)kr * HEADD + kc);
  async_g2l_b128(lds_off32(&vtile[0][vr * 32 + vc]), Vb + (size_t)vr * S_LEN + vc);

  const int NSTEP = S_LEN / 32;            // 64
  for (int st = 0; st < NSTEP - 1; ++st) { // uniform steady-state loop
    const int nxt = (st + 1) & 1;
    const int kn  = st * 32 + 32;
    async_g2l_b128(lds_off32(&ktile[nxt][kr * 64 + kc]),
                   Kb + (size_t)(kn + kr) * HEADD + kc);
    async_g2l_b128(lds_off32(&vtile[nxt][vr * 32 + vc]),
                   Vb + (size_t)vr * S_LEN + kn + vc);
    S_WAIT_ASYNC(0x2);                     // current K+V tiles done (next pair in flight)
    __syncthreads();
    compute(st);
    __syncthreads();                       // all waves done with this K/V buffer
  }
  S_WAIT_ASYNC(0x0);                       // peeled last step
  __syncthreads();
  compute(NSTEP - 1);

  // normalize and store ctx [B, S, H] bf16 (merged heads, row-major for out-proj A-frags)
  float inv[8];
#pragma unroll
  for (int i = 0; i < 8; ++i) inv[i] = 1.0f / lrow[i];
#pragma unroll
  for (int t = 0; t < 4; ++t) {
    const int d = h * HEADD + t * 16 + l16;
#pragma unroll
    for (int i = 0; i < 8; ++i) {
      const int s = q0 + i + half * 8;
      ctx[((size_t)b * S_LEN + s) * HDIM + d] = f2bf(o[t][i] * inv[i]);
    }
  }
}

// ---------------- launch ----------------

extern "C" void kernel_launch(void* const* d_in, const int* in_sizes, int n_in,
                              void* d_out, int out_size, void* d_ws, size_t ws_size,
                              hipStream_t stream) {
  (void)in_sizes; (void)n_in; (void)out_size; (void)ws_size;
  const float* hidden = (const float*)d_in[0];
  const float* amask  = (const float*)d_in[1];
  const float* Wq = (const float*)d_in[2]; const float* bq = (const float*)d_in[3];
  const float* Wk = (const float*)d_in[4]; const float* bk = (const float*)d_in[5];
  const float* Wv = (const float*)d_in[6]; const float* bv = (const float*)d_in[7];
  const float* Wo = (const float*)d_in[8]; const float* bo = (const float*)d_in[9];
  float* out = (float*)d_out;

  char* ws = (char*)d_ws;
  size_t off = 0;
  unsigned short* Xbf = (unsigned short*)(ws + off); off += (size_t)MROWS * HDIM * 2; // 8 MB
  unsigned short* Wqt = (unsigned short*)(ws + off); off += (size_t)HDIM * HDIM * 2;  // 2 MB
  unsigned short* Wkt = (unsigned short*)(ws + off); off += (size_t)HDIM * HDIM * 2;
  unsigned short* Wvt = (unsigned short*)(ws + off); off += (size_t)HDIM * HDIM * 2;
  unsigned short* Wot = (unsigned short*)(ws + off); off += (size_t)HDIM * HDIM * 2;
  unsigned short* Qw  = (unsigned short*)(ws + off); off += (size_t)MROWS * HDIM * 2; // 8 MB
  unsigned short* Kw  = (unsigned short*)(ws + off); off += (size_t)MROWS * HDIM * 2;
  unsigned short* Vtw = (unsigned short*)(ws + off); off += (size_t)MROWS * HDIM * 2;
  unsigned short* Ctx = (unsigned short*)(ws + off); off += (size_t)MROWS * HDIM * 2;
  // total: 48 MB

  dim3 blk(256);
  pack_x_kernel<<<(MROWS * HDIM) / 256, blk, 0, stream>>>(hidden, Xbf, MROWS * HDIM);
  pack_wt_kernel<<<(HDIM * HDIM) / 256, blk, 0, stream>>>(Wq, Wqt);
  pack_wt_kernel<<<(HDIM * HDIM) / 256, blk, 0, stream>>>(Wk, Wkt);
  pack_wt_kernel<<<(HDIM * HDIM) / 256, blk, 0, stream>>>(Wv, Wvt);
  pack_wt_kernel<<<(HDIM * HDIM) / 256, blk, 0, stream>>>(Wo, Wot);

  dim3 gg(HDIM / 64, MROWS / 256);   // (16, 16): 8 waves x (32M x 64N) per block
  gemm_bf16_kernel<<<gg, blk, 0, stream>>>(Xbf, Wqt, bq, (void*)Qw, 0);
  gemm_bf16_kernel<<<gg, blk, 0, stream>>>(Xbf, Wkt, bk, (void*)Kw, 0);
  gemm_bf16_kernel<<<gg, blk, 0, stream>>>(Xbf, Wvt, bv, (void*)Vtw, 1);

  attn_kernel<<<dim3(S_LEN / 16 / 8, BATCH * NHEADS), blk, 0, stream>>>(Qw, Kw, Vtw, amask, Ctx);

  gemm_bf16_kernel<<<gg, blk, 0, stream>>>(Ctx, Wot, bo, (void*)out, 2);
}
